// GraphSAGEEncoder_40303973105857
// MI455X (gfx1250) — compile-verified, matrix operand order checked by reading, not run on hardware
//
#include <hip/hip_runtime.h>
#include <hip/hip_bf16.h>

typedef __bf16 v16bf __attribute__((ext_vector_type(16)));
typedef float  v8f   __attribute__((ext_vector_type(8)));
typedef unsigned int u32;
typedef u32 u32x4 __attribute__((ext_vector_type(4)));
typedef int  v4i   __attribute__((ext_vector_type(4)));
typedef unsigned short ushort_t;

// address-space-qualified pointer types for the async global->LDS builtin
typedef __attribute__((address_space(1))) v4i* gas_v4i_ptr;
typedef __attribute__((address_space(3))) v4i* las_v4i_ptr;

union FragBF {
    u32x4 q[2];
    v16bf v;
};

#if __has_builtin(__builtin_amdgcn_global_load_async_to_lds_b128)
#define USE_ASYNC_LDS 1
#endif

__device__ __forceinline__ unsigned short f2bf(float f) {
    u32 u = __float_as_uint(f);
    u32 r = (u + 0x7FFFu + ((u >> 16) & 1u)) >> 16;   // round-to-nearest-even
    return (unsigned short)r;
}

// copy 16 bytes global -> LDS (async mover when available)
__device__ __forceinline__ void copy_b128(const ushort_t* g, ushort_t* l) {
#ifdef USE_ASYNC_LDS
    __builtin_amdgcn_global_load_async_to_lds_b128(
        (gas_v4i_ptr)g, (las_v4i_ptr)l, 0, 0);
#else
    *(u32x4*)l = *(const u32x4*)g;
#endif
}

__device__ __forceinline__ void copy_wait() {
#ifdef USE_ASYNC_LDS
    asm volatile("s_wait_asynccnt 0x0" ::: "memory");
#endif
}

// ---------------------------------------------------------------------------
// Fused dual GEMM on pre-padded bf16 operands:
//   outL = A @ WtL^T,  outR = A @ WtR^T
//   A   : Mpad x Kpad bf16 (row-major, Kpad % 32 == 0, rows >= gridDim.x*128)
//   Wt* : Hout x Kpad bf16 (column of original W is K-contiguous)
//   out : M x Hout fp32
// Block = 256 threads (8 waves). C tile = 128 x 64. K step 32, double-buffered
// async global->LDS staging overlapped with v_wmma_f32_16x16x32_bf16.
// ---------------------------------------------------------------------------
__global__ void __launch_bounds__(256)
gemm_dual_wmma(const ushort_t* __restrict__ Abf,
               const ushort_t* __restrict__ WtL,
               const ushort_t* __restrict__ WtR,
               float* __restrict__ outL,
               float* __restrict__ outR,
               int M, int Kpad, int Hout) {
    // stride 40 shorts (80 B): 16B-aligned b128 chunks, conflict-free banks
    __shared__ __align__(16) ushort_t sA [2][128 * 40];
    __shared__ __align__(16) ushort_t sBL[2][ 64 * 40];
    __shared__ __align__(16) ushort_t sBR[2][ 64 * 40];

    const int tid  = threadIdx.x;
    const int wave = tid >> 5;
    const int lane = tid & 31;
    const int rowBase = blockIdx.x * 128;
    const int colBase = blockIdx.y * 64;

    const int m     = lane & 15;
    const int khalf = (lane >> 4) * 8;       // ISA 16-bit A/B fragment layout
    const int kIters = Kpad >> 5;

    v8f accL[4], accR[4];
    const v8f vzero = {0.f, 0.f, 0.f, 0.f, 0.f, 0.f, 0.f, 0.f};
#pragma unroll
    for (int i = 0; i < 4; ++i) { accL[i] = vzero; accR[i] = vzero; }

    // stage one 128x32 A tile + two 64x32 B tiles: 1024 b128 chunks total,
    // 4 per thread, no bounds checks (operands are pre-padded).
    auto issue_tile = [&](int it, int buf) {
        const int k0 = it << 5;
#pragma unroll
        for (int h = 0; h < 2; ++h) {            // A: chunks tid, tid+256
            const int c   = tid + h * 256;
            const int row = c >> 2;
            const int q   = (c & 3) * 8;
            copy_b128(Abf + (size_t)(rowBase + row) * Kpad + k0 + q,
                      &sA[buf][row * 40 + q]);
        }
#pragma unroll
        for (int h = 0; h < 2; ++h) {            // B: L then R halves
            const int c = tid + h * 256;
            const int n = (c >> 2) & 63;
            const int q = (c & 3) * 8;
            const ushort_t* g = (h ? WtR : WtL) + (size_t)(colBase + n) * Kpad + k0 + q;
            ushort_t* l = (h ? &sBR[buf][n * 40 + q] : &sBL[buf][n * 40 + q]);
            copy_b128(g, l);
        }
    };

    issue_tile(0, 0);

    for (int it = 0; it < kIters; ++it) {
        const int buf = it & 1;
        copy_wait();                 // this tile's async copies done (this wave)
        __syncthreads();             // every wave's copies done / prev compute done
        if (it + 1 < kIters) issue_tile(it + 1, buf ^ 1);

        FragBF fa;
        {
            const u32x4* pa = (const u32x4*)(&sA[buf][(wave * 16 + m) * 40 + khalf]);
            fa.q[0] = pa[0];         // K = khalf .. khalf+7
            fa.q[1] = pa[2];         // K = khalf+16 .. khalf+23
        }
#pragma unroll
        for (int ct = 0; ct < 4; ++ct) {
            FragBF fbL, fbR;
            const u32x4* pl = (const u32x4*)(&sBL[buf][(ct * 16 + m) * 40 + khalf]);
            const u32x4* pr = (const u32x4*)(&sBR[buf][(ct * 16 + m) * 40 + khalf]);
            fbL.q[0] = pl[0]; fbL.q[1] = pl[2];
            fbR.q[0] = pr[0]; fbR.q[1] = pr[2];
            accL[ct] = __builtin_amdgcn_wmma_f32_16x16x32_bf16(
                false, fa.v, false, fbL.v, (short)0, accL[ct], false, false);
            accR[ct] = __builtin_amdgcn_wmma_f32_16x16x32_bf16(
                false, fa.v, false, fbR.v, (short)0, accR[ct], false, false);
        }
    }

    // C layout: VGPR r -> M = r (lanes 0-15) or r+8 (lanes 16-31), N = lane&15
    const int mrow0 = rowBase + wave * 16 + ((lane < 16) ? 0 : 8);
#pragma unroll
    for (int ct = 0; ct < 4; ++ct) {
        const int col = colBase + ct * 16 + m;
#pragma unroll
        for (int r = 0; r < 8; ++r) {
            const int row = mrow0 + r;
            if (row < M) {
                outL[(size_t)row * Hout + col] = accL[ct][r];
                outR[(size_t)row * Hout + col] = accR[ct][r];
            }
        }
    }
}

// ---------------------------------------------------------------------------
// conversion / helper kernels
// ---------------------------------------------------------------------------
// X [rows x K] fp32 -> out [rowsPad x Kpad] bf16, zero-padded
__global__ void convert_pad_bf16(const float* __restrict__ X, ushort_t* __restrict__ out,
                                 int rows, int K, int Kpad, long long total) {
    long long i = (long long)blockIdx.x * blockDim.x + threadIdx.x;
    if (i >= total) return;
    const int r = (int)(i / Kpad);
    const int k = (int)(i % Kpad);
    float v = (r < rows && k < K) ? X[(size_t)r * K + k] : 0.f;
    out[i] = f2bf(v);
}

// W [K x Hout] fp32 -> out [Hout x Kpad] bf16 (transposed, zero-padded)
__global__ void convert_transpose_bf16(const float* __restrict__ W, ushort_t* __restrict__ out,
                                       int K, int Hout, int Kpad, long long total) {
    long long i = (long long)blockIdx.x * blockDim.x + threadIdx.x;
    if (i >= total) return;
    const int col = (int)(i / Kpad);
    const int k   = (int)(i % Kpad);
    float v = (k < K) ? W[(size_t)k * Hout + col] : 0.f;
    out[i] = f2bf(v);
}

__global__ void zero_f32(float* p, long long n) {
    long long i = (long long)blockIdx.x * blockDim.x + threadIdx.x;
    if (i < n) p[i] = 0.f;
}

__global__ void count_deg(const int* __restrict__ dst, float* __restrict__ cnt, int E) {
    int i = blockIdx.x * blockDim.x + threadIdx.x;
    if (i < E) atomicAdd(&cnt[dst[i]], 1.0f);
}

// agg[dst] += y[src] over F features, 4 floats per thread
__global__ void scatter_add(const float* __restrict__ y,
                            const int* __restrict__ src,
                            const int* __restrict__ dst,
                            float* __restrict__ agg,
                            int F, long long total) {
    long long i = (long long)blockIdx.x * blockDim.x + threadIdx.x;
    if (i >= total) return;
    const int chunks = F >> 2;
    const int e = (int)(i / chunks);
    const int c = (int)(i % chunks) * 4;
    const int s = src[e], d = dst[e];
    float4 v = *(const float4*)(y + (size_t)s * F + c);
    float* o = agg + (size_t)d * F + c;
    atomicAdd(o + 0, v.x);
    atomicAdd(o + 1, v.y);
    atomicAdd(o + 2, v.z);
    atomicAdd(o + 3, v.w);
}

// hbf = bf16( relu( BN( agg/deg + b1 + yR ) ) )   -- layer-2 GEMM A operand
__global__ void epilogue_bn_relu(const float* __restrict__ agg,
                                 const float* __restrict__ yR,
                                 const float* __restrict__ cnt,
                                 const float* __restrict__ b1,
                                 const float* __restrict__ gamma,
                                 const float* __restrict__ beta,
                                 const float* __restrict__ mean,
                                 const float* __restrict__ var,
                                 ushort_t* __restrict__ hbf,
                                 int H, long long total) {
    long long i = (long long)blockIdx.x * blockDim.x + threadIdx.x;
    if (i >= total) return;
    const int f = (int)(i % H);
    const int n = (int)(i / H);
    float deg = fmaxf(cnt[n], 1.0f);
    float v = agg[i] / deg + b1[f] + yR[i];
    float s = gamma[f] * __frsqrt_rn(var[f] + 1e-5f);
    v = (v - mean[f]) * s + beta[f];
    hbf[i] = f2bf(fmaxf(v, 0.0f));
}

// z = agg/deg + b2 + yR
__global__ void final_combine(const float* __restrict__ agg,
                              const float* __restrict__ yR,
                              const float* __restrict__ cnt,
                              const float* __restrict__ b2,
                              float* __restrict__ out,
                              int Z, long long total) {
    long long i = (long long)blockIdx.x * blockDim.x + threadIdx.x;
    if (i >= total) return;
    const int f = (int)(i % Z);
    const int n = (int)(i / Z);
    float deg = fmaxf(cnt[n], 1.0f);
    out[i] = agg[i] / deg + b2[f] + yR[i];
}

// ---------------------------------------------------------------------------
extern "C" void kernel_launch(void* const* d_in, const int* in_sizes, int n_in,
                              void* d_out, int out_size, void* d_ws, size_t ws_size,
                              hipStream_t stream) {
    const float* x   = (const float*)d_in[0];
    const int*  eidx = (const int*)d_in[1];
    const float* W1l = (const float*)d_in[2];
    const float* b1  = (const float*)d_in[3];
    const float* W1r = (const float*)d_in[4];
    const float* gam = (const float*)d_in[5];
    const float* bet = (const float*)d_in[6];
    const float* mea = (const float*)d_in[7];
    const float* var = (const float*)d_in[8];
    const float* W2l = (const float*)d_in[9];
    const float* b2  = (const float*)d_in[10];
    const float* W2r = (const float*)d_in[11];
    float* out = (float*)d_out;

    const int H = in_sizes[3];             // 512
    const int Z = in_sizes[10];            // 64
    const int G = in_sizes[2] / H;         // 2000
    const int N = in_sizes[0] / G;         // 50000
    const int E = in_sizes[1] / 2;         // 800000
    const int Kpad1 = ((G + 31) / 32) * 32;       // 2016
    const int Npad  = ((N + 127) / 128) * 128;    // 50048
    const int* src = eidx;
    const int* dst = eidx + E;

    // workspace layout
    const size_t NH = (size_t)N * H;
    float* ws   = (float*)d_ws;
    float* cnt  = ws;                                  // N
    float* y1l  = cnt + N;                             // N*H
    float* y1r  = y1l + NH;                            // N*H
    float* agg1 = y1r + NH;                            // N*H
    ushort_t* xbf  = (ushort_t*)(agg1 + NH);           // Npad*Kpad1
    ushort_t* hbf  = xbf + (size_t)Npad * Kpad1;       // Npad*H
    ushort_t* w1lt = hbf + (size_t)Npad * H;           // H*Kpad1
    ushort_t* w1rt = w1lt + (size_t)H * Kpad1;         // H*Kpad1
    ushort_t* w2lt = w1rt + (size_t)H * Kpad1;         // Z*H
    ushort_t* w2rt = w2lt + (size_t)Z * H;             // Z*H
    // phase-2 fp32 buffers reuse the y1l region
    float* y2l  = y1l;
    float* y2r  = y1l + (size_t)N * Z;
    float* agg2 = y1l + 2 * (size_t)N * Z;

    const int T = 256;
    auto blocks = [](long long n, int t) { return (int)((n + t - 1) / t); };

    // ---- pre-convert operands to padded bf16 (guard-free async GEMM) ----
    long long txa = (long long)Npad * Kpad1;
    convert_pad_bf16<<<blocks(txa, T), T, 0, stream>>>(x, xbf, N, G, Kpad1, txa);
    long long tw1 = (long long)H * Kpad1;
    convert_transpose_bf16<<<blocks(tw1, T), T, 0, stream>>>(W1l, w1lt, G, H, Kpad1, tw1);
    convert_transpose_bf16<<<blocks(tw1, T), T, 0, stream>>>(W1r, w1rt, G, H, Kpad1, tw1);
    long long tw2 = (long long)Z * H;
    convert_transpose_bf16<<<blocks(tw2, T), T, 0, stream>>>(W2l, w2lt, H, Z, H, tw2);
    convert_transpose_bf16<<<blocks(tw2, T), T, 0, stream>>>(W2r, w2rt, H, Z, H, tw2);

    // ---- degree counts + zero agg1 ----
    zero_f32<<<blocks(N, T), T, 0, stream>>>(cnt, N);
    zero_f32<<<blocks((long long)NH, T), T, 0, stream>>>(agg1, (long long)NH);
    count_deg<<<blocks(E, T), T, 0, stream>>>(dst, cnt, E);

    // ---- layer 1: project (WMMA) then aggregate (aggregation is linear) ----
    dim3 g1(Npad / 128, H / 64);
    gemm_dual_wmma<<<g1, T, 0, stream>>>(xbf, w1lt, w1rt, y1l, y1r, N, Kpad1, H);

    long long tot1 = (long long)E * (H / 4);
    scatter_add<<<blocks(tot1, T), T, 0, stream>>>(y1l, src, dst, agg1, H, tot1);

    epilogue_bn_relu<<<blocks((long long)NH, T), T, 0, stream>>>(
        agg1, y1r, cnt, b1, gam, bet, mea, var, hbf, H, (long long)NH);
    if (Npad > N) {  // zero bf16 pad rows of layer-2 A operand
        long long padf = (long long)(Npad - N) * H / 2;
        zero_f32<<<blocks(padf, T), T, 0, stream>>>((float*)(hbf + NH), padf);
    }

    // ---- layer 2 ----
    zero_f32<<<blocks((long long)N * Z, T), T, 0, stream>>>(agg2, (long long)N * Z);
    dim3 g2(Npad / 128, Z / 64);
    gemm_dual_wmma<<<g2, T, 0, stream>>>(hbf, w2lt, w2rt, y2l, y2r, N, H, Z);

    long long tot2 = (long long)E * (Z / 4);
    scatter_add<<<blocks(tot2, T), T, 0, stream>>>(y2l, src, dst, agg2, Z, tot2);

    long long nz = (long long)N * Z;
    final_combine<<<blocks(nz, T), T, 0, stream>>>(agg2, y2r, cnt, b2, out, Z, nz);
}
